// IGNN_57964878627399
// MI455X (gfx1250) — matrix operand order repaired and checked by reading, not compiled, not run on hardware
//
#include <hip/hip_runtime.h>
#include <hip/hip_bf16.h>
#include <math.h>

#define N_NODES 50000
#define N_EDGES 800000
#define DD      128

typedef __attribute__((ext_vector_type(16))) __bf16 v16bf;
typedef __attribute__((ext_vector_type(8)))  float  v8f;

union BF16x16 {
    v16bf  v;
    __bf16 h[16];
    uint4  q[2];
};

__device__ __forceinline__ v8f wmma_bf16(v16bf a, v16bf b, v8f c) {
    return __builtin_amdgcn_wmma_f32_16x16x32_bf16(false, a, false, b, (short)0, c, false, false);
}

// Packed B-fragment layout: fragment f = kt*8+nt, lane l -> 16 contiguous bf16 (32B).
__device__ __forceinline__ v16bf load_bfrag(const __bf16* __restrict__ Wp, int kt, int nt, int lane) {
    return *(const v16bf*)(Wp + ((((size_t)kt * 8 + nt) * 32 + (size_t)lane) * 16));
}

__device__ __forceinline__ void cvt8(BF16x16& A, int base, float4 a, float4 b) {
    A.h[base + 0] = (__bf16)a.x; A.h[base + 1] = (__bf16)a.y;
    A.h[base + 2] = (__bf16)a.z; A.h[base + 3] = (__bf16)a.w;
    A.h[base + 4] = (__bf16)b.x; A.h[base + 5] = (__bf16)b.y;
    A.h[base + 6] = (__bf16)b.z; A.h[base + 7] = (__bf16)b.w;
}

// A fragment for one 16x32 bf16 K-tile from a row-major f32 row.
__device__ __forceinline__ void loadA(BF16x16& A, const float* __restrict__ row, int kbase) {
    const float4 a0 = *(const float4*)(row + kbase);
    const float4 a1 = *(const float4*)(row + kbase + 4);
    const float4 b0 = *(const float4*)(row + kbase + 16);
    const float4 b1 = *(const float4*)(row + kbase + 20);
    cvt8(A, 0, a0, a1);
    cvt8(A, 8, b0, b1);
}

// v_rcp_f32-based SiLU (avoids the IEEE divide expansion).
__device__ __forceinline__ float silu_f(float x) {
    return x * __builtin_amdgcn_rcpf(1.0f + __expf(-x));
}

// Wave-private LDS store->load ordering (no cross-wave sharing, so no block barrier).
__device__ __forceinline__ void wave_lds_fence() {
    asm volatile("s_wait_dscnt 0" ::: "memory");
}

// ---------------------------------------------------------------------------
// Pack a [K x 128] row-major f32 weight matrix into bf16 WMMA B fragments.
// B fragment (16x16x32 bf16): lane<16 -> column n=lane, K kbase+0..7,16..23;
//                             lane>=16 -> column n=lane-16, K kbase+8..15,24..31.
// Rows >= Kvalid are zero padding.
// ---------------------------------------------------------------------------
__global__ void pack_weights_kernel(const float* __restrict__ W, int Kvalid, int ktiles,
                                    __bf16* __restrict__ out) {
    const int total = ktiles * 256;  // 8 ntiles * 32 lanes per ktile
    for (int c = blockIdx.x * blockDim.x + threadIdx.x; c < total; c += gridDim.x * blockDim.x) {
        const int lane = c & 31;
        const int nt   = (c >> 5) & 7;
        const int kt   = c >> 8;
        const int n    = nt * 16 + (lane & 15);
        const int kb   = kt * 32 + ((lane >= 16) ? 8 : 0);
        __bf16* dst = out + (size_t)c * 16;
        #pragma unroll
        for (int kk = 0; kk < 16; ++kk) {
            const int k = kb + kk + ((kk >= 8) ? 8 : 0);
            const float v = (k < Kvalid) ? W[(size_t)k * DD + n] : 0.0f;
            dst[kk] = (__bf16)v;
        }
    }
}

__global__ void zero_f32_kernel(float* __restrict__ p, long long n) {
    const long long i = (long long)blockIdx.x * blockDim.x + threadIdx.x;
    if (i < n) p[i] = 0.0f;
}

// ---------------------------------------------------------------------------
// Edge kernel: one wave per 32 edges (two 16-row A tiles share each B fragment,
// halving weight-fragment cache traffic per edge vs 16 edges/wave).
//   h   = silu(edge_in @ We1 + be1)   edge_in = [X[src] | X[dst] | dist | emb_e] (K=259 -> 9 K-tiles)
//   mij = silu(h @ We2 + be2)                                                   (K=128 -> 4 K-tiles)
//   mij -> global (nontemporal); scatter-add mij into mi[dst] with f32 atomics.
// ---------------------------------------------------------------------------
__global__ __launch_bounds__(256)
void edge_kernel(const float* __restrict__ X, const float* __restrict__ embN,
                 const float* __restrict__ embE, const long long* __restrict__ ei,
                 const __bf16* __restrict__ We1p, const float* __restrict__ be1,
                 const __bf16* __restrict__ We2p, const float* __restrict__ be2,
                 float* __restrict__ mij_out, float* __restrict__ mi) {
    __shared__ __align__(16) __bf16 hbuf[8][32 * DD];   // 8 waves x 8KB = 64KB

    const int lane = threadIdx.x & 31;
    const int wave = threadIdx.x >> 5;
    const int e0   = (blockIdx.x * 8 + wave) * 32;
    const int lo   = lane & 15;
    const bool hi  = lane >= 16;

    // This lane's A-matrix rows: edge eA (tile0, rows 0..15) and eB (tile1, rows 16..31).
    const int eA = e0 + lo;
    const int eB = e0 + 16 + lo;
    const int srcA = (int)ei[eA], dstA = (int)ei[N_EDGES + eA];
    const int srcB = (int)ei[eB], dstB = (int)ei[N_EDGES + eB];
    const float* __restrict__ sA = X + (size_t)srcA * DD;
    const float* __restrict__ dA = X + (size_t)dstA * DD;
    const float* __restrict__ sB = X + (size_t)srcB * DD;
    const float* __restrict__ dB = X + (size_t)dstB * DD;

    // Extra features (K = 256,257,258) for both edges.
    float fA0, fA1, fA2, fB0, fB1, fB2;
    {
        const float* a = embN + (size_t)srcA * 3;
        const float* b = embN + (size_t)dstA * 3;
        const float dx = b[0] - a[0], dy = b[1] - a[1], dz = b[2] - a[2];
        fA0 = __builtin_amdgcn_sqrtf(dx * dx + dy * dy + dz * dz);
        fA1 = embE[(size_t)eA * 2 + 0];
        fA2 = embE[(size_t)eA * 2 + 1];
    }
    {
        const float* a = embN + (size_t)srcB * 3;
        const float* b = embN + (size_t)dstB * 3;
        const float dx = b[0] - a[0], dy = b[1] - a[1], dz = b[2] - a[2];
        fB0 = __builtin_amdgcn_sqrtf(dx * dx + dy * dy + dz * dz);
        fB1 = embE[(size_t)eB * 2 + 0];
        fB2 = embE[(size_t)eB * 2 + 1];
    }

    v8f acc0[8] = {}, acc1[8] = {};

    // GEMM1: K-tiles 0..3 from X[src], 4..7 from X[dst]; each B fragment feeds 2 WMMAs.
    for (int kt = 0; kt < 8; ++kt) {
        const float* __restrict__ r0 = (kt < 4) ? sA : dA;
        const float* __restrict__ r1 = (kt < 4) ? sB : dB;
        const int kbase = (kt & 3) * 32 + (hi ? 8 : 0);
        BF16x16 A0, A1;
        loadA(A0, r0, kbase);
        loadA(A1, r1, kbase);
        #pragma unroll
        for (int nt = 0; nt < 8; ++nt) {
            const v16bf B = load_bfrag(We1p, kt, nt, lane);
            acc0[nt] = wmma_bf16(A0.v, B, acc0[nt]);
            acc1[nt] = wmma_bf16(A1.v, B, acc1[nt]);
        }
    }
    // K-tile 8: [dist, e0, e1, 0...]; only lanes<16 (K 256..263) carry data.
    {
        BF16x16 A0, A1;
        #pragma unroll
        for (int i = 0; i < 16; ++i) { A0.h[i] = (__bf16)0.0f; A1.h[i] = (__bf16)0.0f; }
        if (!hi) {
            A0.h[0] = (__bf16)fA0; A0.h[1] = (__bf16)fA1; A0.h[2] = (__bf16)fA2;
            A1.h[0] = (__bf16)fB0; A1.h[1] = (__bf16)fB1; A1.h[2] = (__bf16)fB2;
        }
        #pragma unroll
        for (int nt = 0; nt < 8; ++nt) {
            const v16bf B = load_bfrag(We1p, 8, nt, lane);
            acc0[nt] = wmma_bf16(A0.v, B, acc0[nt]);
            acc1[nt] = wmma_bf16(A1.v, B, acc1[nt]);
        }
    }

    // Bias + SiLU, restage h (bf16) into this wave's LDS tile (C-layout -> row-major).
    float bias1[8];
    #pragma unroll
    for (int nt = 0; nt < 8; ++nt) bias1[nt] = be1[nt * 16 + lo];

    __bf16* hw = hbuf[wave];
    #pragma unroll
    for (int nt = 0; nt < 8; ++nt) {
        const int col = nt * 16 + lo;
        #pragma unroll
        for (int r = 0; r < 8; ++r) {
            const int M = r + (hi ? 8 : 0);
            hw[M * DD + col]        = (__bf16)silu_f(acc0[nt][r] + bias1[nt]);
            hw[(16 + M) * DD + col] = (__bf16)silu_f(acc1[nt][r] + bias1[nt]);
        }
    }
    wave_lds_fence();

    // GEMM2: A fragments read back from LDS in A-layout (two 16B chunks per lane per tile).
    v8f c0[8] = {}, c1[8] = {};
    #pragma unroll
    for (int kt = 0; kt < 4; ++kt) {
        const int kbase = kt * 32 + (hi ? 8 : 0);
        BF16x16 A0, A1;
        A0.q[0] = *(const uint4*)(hw + lo * DD + kbase);
        A0.q[1] = *(const uint4*)(hw + lo * DD + kbase + 16);
        A1.q[0] = *(const uint4*)(hw + (16 + lo) * DD + kbase);
        A1.q[1] = *(const uint4*)(hw + (16 + lo) * DD + kbase + 16);
        #pragma unroll
        for (int nt = 0; nt < 8; ++nt) {
            const v16bf B = load_bfrag(We2p, kt, nt, lane);
            c0[nt] = wmma_bf16(A0.v, B, c0[nt]);
            c1[nt] = wmma_bf16(A1.v, B, c1[nt]);
        }
    }

    float bias2[8];
    #pragma unroll
    for (int nt = 0; nt < 8; ++nt) bias2[nt] = be2[nt * 16 + lo];

    #pragma unroll
    for (int t = 0; t < 2; ++t) {
        #pragma unroll
        for (int r = 0; r < 8; ++r) {
            const int M = r + (hi ? 8 : 0);
            const int e = e0 + t * 16 + M;
            const int d = (int)ei[N_EDGES + e];
            float* __restrict__ miRow  = mi + (size_t)d * DD;
            float* __restrict__ outRow = mij_out + (size_t)e * DD;
            #pragma unroll
            for (int nt = 0; nt < 8; ++nt) {
                const int col = nt * 16 + lo;
                const float v = (t == 0) ? c0[nt][r] : c1[nt][r];
                const float s = silu_f(v + bias2[nt]);
                __builtin_nontemporal_store(s, outRow + col);   // streaming 409MB output
                unsafeAtomicAdd(miRow + col, s);                // global_atomic_add_f32
            }
        }
    }
}

// ---------------------------------------------------------------------------
// Node kernel: one wave per 16 nodes.
//   X_out = silu([X | mi] @ Wh1 + bh1) @ Wh2 + bh2   (K=256 then K=128)
// ---------------------------------------------------------------------------
__global__ __launch_bounds__(256)
void node_kernel(const float* __restrict__ X, const float* __restrict__ mi,
                 const __bf16* __restrict__ Wh1p, const float* __restrict__ bh1,
                 const __bf16* __restrict__ Wh2p, const float* __restrict__ bh2,
                 float* __restrict__ Xout) {
    __shared__ __align__(16) __bf16 hbuf[8][16 * DD];

    const int lane = threadIdx.x & 31;
    const int wave = threadIdx.x >> 5;
    const int tile = blockIdx.x * 8 + wave;
    if (tile >= N_NODES / 16) return;   // no block barriers used -> safe early exit
    const int m0 = tile * 16;
    const int lo = lane & 15;
    const bool hi = lane >= 16;

    const float* __restrict__ xRow = X  + (size_t)(m0 + lo) * DD;
    const float* __restrict__ mRow = mi + (size_t)(m0 + lo) * DD;

    v8f acc[8] = {};
    for (int kt = 0; kt < 8; ++kt) {
        const float* __restrict__ row = (kt < 4) ? xRow : mRow;
        const int kbase = (kt & 3) * 32 + (hi ? 8 : 0);
        BF16x16 A;
        loadA(A, row, kbase);
        #pragma unroll
        for (int nt = 0; nt < 8; ++nt)
            acc[nt] = wmma_bf16(A.v, load_bfrag(Wh1p, kt, nt, lane), acc[nt]);
    }

    float bias1[8];
    #pragma unroll
    for (int nt = 0; nt < 8; ++nt) bias1[nt] = bh1[nt * 16 + lo];

    __bf16* hw = hbuf[wave];
    #pragma unroll
    for (int nt = 0; nt < 8; ++nt) {
        const int col = nt * 16 + lo;
        #pragma unroll
        for (int r = 0; r < 8; ++r) {
            const int M = r + (hi ? 8 : 0);
            hw[M * DD + col] = (__bf16)silu_f(acc[nt][r] + bias1[nt]);
        }
    }
    wave_lds_fence();

    v8f acc2[8] = {};
    #pragma unroll
    for (int kt = 0; kt < 4; ++kt) {
        const int kbase = kt * 32 + (hi ? 8 : 0);
        BF16x16 A;
        A.q[0] = *(const uint4*)(hw + lo * DD + kbase);
        A.q[1] = *(const uint4*)(hw + lo * DD + kbase + 16);
        #pragma unroll
        for (int nt = 0; nt < 8; ++nt)
            acc2[nt] = wmma_bf16(A.v, load_bfrag(Wh2p, kt, nt, lane), acc2[nt]);
    }

    float bias2[8];
    #pragma unroll
    for (int nt = 0; nt < 8; ++nt) bias2[nt] = bh2[nt * 16 + lo];

    #pragma unroll
    for (int r = 0; r < 8; ++r) {
        const int M = r + (hi ? 8 : 0);
        float* __restrict__ outRow = Xout + (size_t)(m0 + M) * DD;
        #pragma unroll
        for (int nt = 0; nt < 8; ++nt) {
            const int col = nt * 16 + lo;
            __builtin_nontemporal_store(acc2[nt][r] + bias2[nt], outRow + col);
        }
    }
}

// ---------------------------------------------------------------------------

extern "C" void kernel_launch(void* const* d_in, const int* in_sizes, int n_in,
                              void* d_out, int out_size, void* d_ws, size_t ws_size,
                              hipStream_t stream) {
    const float*     X    = (const float*)d_in[0];
    /* d_in[1] = E (unused; reference replaces it with mij) */
    const float*     embN = (const float*)d_in[2];
    const float*     embE = (const float*)d_in[3];
    const long long* ei   = (const long long*)d_in[4];
    const float*     We1  = (const float*)d_in[5];
    const float*     be1  = (const float*)d_in[6];
    const float*     We2  = (const float*)d_in[7];
    const float*     be2  = (const float*)d_in[8];
    const float*     Wh1  = (const float*)d_in[9];
    const float*     bh1  = (const float*)d_in[10];
    const float*     Wh2  = (const float*)d_in[11];
    const float*     bh2  = (const float*)d_in[12];

    // Output layout (flat f32): X_out | mij | emb_nodes | emb_edges
    float* out      = (float*)d_out;
    float* Xout     = out;
    float* mij      = Xout + (size_t)N_NODES * DD;
    float* embN_out = mij + (size_t)N_EDGES * DD;
    float* embE_out = embN_out + (size_t)N_NODES * 3;

    // Workspace: mi accumulator (f32) then packed bf16 weight fragments.
    const long long miElems = (long long)N_NODES * DD;  // 6.4M floats
    float*  mi   = (float*)d_ws;
    __bf16* We1p = (__bf16*)((char*)d_ws + (size_t)miElems * 4);
    __bf16* We2p = We1p + (size_t)9 * 8 * 512;   // 9 K-tiles (K=288, 259 valid)
    __bf16* Wh1p = We2p + (size_t)4 * 8 * 512;   // 4 K-tiles (K=128)
    __bf16* Wh2p = Wh1p + (size_t)8 * 8 * 512;   // 8 K-tiles (K=256)

    // 1) zero mi + pack weights to bf16 fragment layout (L2-resident thereafter)
    zero_f32_kernel<<<(int)((miElems + 255) / 256), 256, 0, stream>>>(mi, miElems);
    pack_weights_kernel<<<9, 256, 0, stream>>>(We1, 2 * DD + 3, 9, We1p);
    pack_weights_kernel<<<4, 256, 0, stream>>>(We2, DD, 4, We2p);
    pack_weights_kernel<<<8, 256, 0, stream>>>(Wh1, 2 * DD, 8, Wh1p);
    pack_weights_kernel<<<4, 256, 0, stream>>>(Wh2, DD, 4, Wh2p);

    // 2) edge MLP + scatter-add (8 waves/block * 32 edges/wave = 256 edges/block)
    edge_kernel<<<N_EDGES / 256, 256, 0, stream>>>(X, embN, embE, ei,
                                                   We1p, be1, We2p, be2, mij, mi);

    // 3) node MLP (3125 wave-tiles -> 391 blocks of 8 waves)
    node_kernel<<<(N_NODES / 16 + 7) / 8, 256, 0, stream>>>(X, mi, Wh1p, bh1, Wh2p, bh2, Xout);

    // 4) passthrough outputs
    hipMemcpyAsync(embN_out, embN, (size_t)N_NODES * 3 * sizeof(float),
                   hipMemcpyDeviceToDevice, stream);
    hipMemcpyAsync(embE_out, embE, (size_t)N_EDGES * 2 * sizeof(float),
                   hipMemcpyDeviceToDevice, stream);
}
